// PixelPatchTransformerFlow_81784767250919
// MI455X (gfx1250) — matrix operand order repaired and checked by reading, not compiled, not run on hardware
//
#include <hip/hip_runtime.h>
#include <hip/hip_bf16.h>

// ---------------------------------------------------------------------------
// CDNA5 (gfx1250) implementation of PixelPatchTransformerFlow.
// All GEMMs (QKV/O projections, FFN, encoder, cross-attn V path) and both
// attention matmuls run on v_wmma_f32_16x16x32_bf16 (wave32, f32 accum).
// Exact simplifications used: encoder seq-len==1  => attention == V;
// decoder cross-attention memory seq-len==1       => output == V(memory).
// ---------------------------------------------------------------------------

typedef __attribute__((ext_vector_type(16))) __bf16 v16bf;
typedef __attribute__((ext_vector_type(8)))  __bf16 v8bf;
typedef __attribute__((ext_vector_type(8)))  float  v8f;

__device__ __forceinline__ __bf16 f2bf(float f) {
    unsigned u = __builtin_bit_cast(unsigned, f);
    unsigned r = (u + 0x7fffu + ((u >> 16) & 1u)) >> 16;
    unsigned short h = (unsigned short)r;
    return __builtin_bit_cast(__bf16, h);
}
__device__ __forceinline__ float bf2f(__bf16 h) {
    unsigned short s = __builtin_bit_cast(unsigned short, h);
    unsigned u = ((unsigned)s) << 16;
    return __builtin_bit_cast(float, u);
}

__device__ __forceinline__ v16bf cat16(v8bf a, v8bf b) {
    return __builtin_shufflevector(a, b, 0,1,2,3,4,5,6,7,8,9,10,11,12,13,14,15);
}

// A fragment: 16x32 bf16 tile, row-major source with leading dim ld.
// Lane L (L<16): M=L, K={0..7,16..23}; L>=16: M=L-16, K={8..15,24..31}.
__device__ __forceinline__ v16bf load_frag_a(const __bf16* A, int ld, int lane) {
    int m  = lane & 15;
    int hi = (lane >> 4) & 1;
    const __bf16* p = A + (size_t)m * ld + hi * 8;
    v8bf a0 = *reinterpret_cast<const v8bf*>(p);
    v8bf a1 = *reinterpret_cast<const v8bf*>(p + 16);
    return cat16(a0, a1);
}

// B fragment: 32x16 bf16 tile taken from Wt[N][K] (column-of-W contiguous).
// Lane L: N=L&15, K={0..7,16..23} (L<16) / {8..15,24..31} (L>=16).
__device__ __forceinline__ v16bf load_frag_b(const __bf16* Bp, int ld, int lane) {
    int n  = lane & 15;
    int hi = (lane >> 4) & 1;
    const __bf16* p = Bp + (size_t)n * ld + hi * 8;
    v8bf b0 = *reinterpret_cast<const v8bf*>(p);
    v8bf b1 = *reinterpret_cast<const v8bf*>(p + 16);
    return cat16(b0, b1);
}

__device__ __forceinline__ v8f wmma_bf16(v16bf a, v16bf b, v8f c) {
    return __builtin_amdgcn_wmma_f32_16x16x32_bf16(false, a, false, b, (short)0, c, false, false);
}

// ---------------------------------------------------------------------------
// GEMM: C[M x N] = act(A[M x K](bf16) @ W[K x N] + bias),  W given as Wt[N][K]
// block = 128 threads (4 waves), block tile 128x64, wave tile 32x64
// (2 A-frags x 4 B-frags, each B-frag reused twice). ~80 VGPRs, no spills.
// ---------------------------------------------------------------------------
template<bool RELU, bool RESID, bool BF16OUT>
__global__ void gemm_kernel(const __bf16* __restrict__ A,
                            const __bf16* __restrict__ Wt,
                            const float* __restrict__ bias,
                            float* __restrict__ Cf,
                            __bf16* __restrict__ Cb,
                            int ldc,
                            const float* __restrict__ resid,
                            int K) {
    const int lane = threadIdx.x & 31;
    const int wave = threadIdx.x >> 5;
    const int row0 = blockIdx.x * 128 + wave * 32;
    const int col0 = blockIdx.y * 64;

    v8f acc[2][4];
    #pragma unroll
    for (int mi = 0; mi < 2; ++mi)
        #pragma unroll
        for (int i = 0; i < 4; ++i)
            #pragma unroll
            for (int j = 0; j < 8; ++j) acc[mi][i][j] = 0.f;

    const __bf16* Arow0 = A + (size_t)row0 * K;
    const __bf16* Arow1 = A + (size_t)(row0 + 16) * K;
    for (int k0 = 0; k0 < K; k0 += 32) {
        v16bf a0 = load_frag_a(Arow0 + k0, K, lane);
        v16bf a1 = load_frag_a(Arow1 + k0, K, lane);
        #pragma unroll
        for (int nf = 0; nf < 4; ++nf) {
            v16bf b = load_frag_b(Wt + (size_t)(col0 + nf * 16) * K + k0, K, lane);
            acc[0][nf] = wmma_bf16(a0, b, acc[0][nf]);
            acc[1][nf] = wmma_bf16(a1, b, acc[1][nf]);
        }
    }

    const int n = lane & 15;
    const int mbase = (lane >> 4) * 8;
    #pragma unroll
    for (int mi = 0; mi < 2; ++mi) {
        #pragma unroll
        for (int nf = 0; nf < 4; ++nf) {
            int c = col0 + nf * 16 + n;
            float bb = bias[c];
            #pragma unroll
            for (int v = 0; v < 8; ++v) {
                size_t idx = (size_t)(row0 + mi * 16 + mbase + v) * ldc + c;
                float val = acc[mi][nf][v] + bb;
                if (RESID) val += resid[idx];
                if (RELU)  val = fmaxf(val, 0.f);
                if (BF16OUT) Cb[idx] = f2bf(val);
                else         Cf[idx] = val;
            }
        }
    }
}

// ---------------------------------------------------------------------------
// Fused attention: one block per (q-tile-of-64, head, batch).
// blockDim 128 (4 waves, 16 query rows each). S=256, dh=64, scale=1/8.
// Scores are streamed tile-by-tile into a per-wave f32 LDS slab (only ONE
// 8-VGPR accumulator live at a time -> no scratch spills), softmax is done
// LDS-resident, normalized P written bf16 and fed back as WMMA A-fragments.
// LDS: K[256][64] + Vt[64][256] (bf16) + S[4][16][256] f32 + P[4][16][256]
//      = 32 + 32 + 64 + 32 = 160KB dynamic.
// ---------------------------------------------------------------------------
__global__ void attn_kernel(const __bf16* __restrict__ qkv,   // [32768][1536]
                            __bf16* __restrict__ out) {       // [32768][512]
    extern __shared__ char smem[];
    __bf16* k_lds  = (__bf16*)smem;                      // [256][64]
    __bf16* vt_lds = k_lds + 256 * 64;                   // [64][256]
    float*  s_lds  = (float*)(vt_lds + 64 * 256);        // [4][16][256]
    __bf16* p_lds  = (__bf16*)(s_lds + 4 * 16 * 256);    // [4][16][256]

    const int tid  = threadIdx.x;
    const int lane = tid & 31;
    const int wave = tid >> 5;
    const int qt   = blockIdx.x;
    const int h    = blockIdx.y;
    const int pb   = blockIdx.z;
    const size_t base = (size_t)pb * 256 * 1536;

    for (int idx = tid; idx < 256 * 64; idx += 128) {
        int s = idx >> 6, d = idx & 63;
        k_lds[idx]          = qkv[base + (size_t)s * 1536 + 512  + h * 64 + d];
        vt_lds[d * 256 + s] = qkv[base + (size_t)s * 1536 + 1024 + h * 64 + d];
    }
    __syncthreads();

    const int q0 = qt * 64 + wave * 16;
    const __bf16* qptr = qkv + base + (size_t)q0 * 1536 + h * 64;
    const int n = lane & 15;
    const int mbase = (lane >> 4) * 8;
    float* sw = s_lds + wave * 16 * 256;   // this wave's 16x256 score slab
    __bf16* pw = p_lds + wave * 16 * 256;

    // scores = (Q K^T) * 1/8, streamed tile-by-tile into LDS
    const float scale = 0.125f;  // 1/sqrt(64)
    v16bf qa0 = load_frag_a(qptr, 1536, lane);
    v16bf qa1 = load_frag_a(qptr + 32, 1536, lane);
    #pragma unroll
    for (int nf = 0; nf < 16; ++nf) {
        v8f acc;
        #pragma unroll
        for (int j = 0; j < 8; ++j) acc[j] = 0.f;
        v16bf b0 = load_frag_b(k_lds + (size_t)(nf * 16) * 64, 64, lane);
        acc = wmma_bf16(qa0, b0, acc);
        v16bf b1 = load_frag_b(k_lds + (size_t)(nf * 16) * 64 + 32, 64, lane);
        acc = wmma_bf16(qa1, b1, acc);
        #pragma unroll
        for (int v = 0; v < 8; ++v)
            sw[(mbase + v) * 256 + nf * 16 + n] = acc[v] * scale;
    }
    // (per-wave slab: DS ops from one wave are in-order; no barrier needed)

    // row softmax on LDS: lane pair (L, L^16) owns row L&15, 128 cols each
    {
        float* srow = sw + (lane & 15) * 256;
        const int c0 = (lane >> 4) * 128;
        float mx = -3.0e38f;
        for (int c = 0; c < 128; ++c) mx = fmaxf(mx, srow[c0 + c]);
        mx = fmaxf(mx, __shfl_xor(mx, 16));
        float sum = 0.f;
        for (int c = 0; c < 128; ++c) {
            float e = __expf(srow[c0 + c] - mx);
            srow[c0 + c] = e;
            sum += e;
        }
        sum += __shfl_xor(sum, 16);
        float inv = 1.f / sum;
        __bf16* prow = pw + (lane & 15) * 256;
        for (int c = 0; c < 128; ++c)
            prow[c0 + c] = f2bf(srow[c0 + c] * inv);
    }

    // out(16x64) = P(16x256) @ V(256x64)
    v8f oacc[4];
    #pragma unroll
    for (int i = 0; i < 4; ++i)
        #pragma unroll
        for (int j = 0; j < 8; ++j) oacc[i][j] = 0.f;
    for (int k0 = 0; k0 < 256; k0 += 32) {
        v16bf a = load_frag_a(pw + k0, 256, lane);
        #pragma unroll
        for (int nf = 0; nf < 4; ++nf) {
            v16bf b = load_frag_b(vt_lds + (size_t)(nf * 16) * 256 + k0, 256, lane);
            oacc[nf] = wmma_bf16(a, b, oacc[nf]);
        }
    }
    #pragma unroll
    for (int nf = 0; nf < 4; ++nf)
        #pragma unroll
        for (int v = 0; v < 8; ++v)
            out[(size_t)(pb * 256 + q0 + mbase + v) * 512 + h * 64 + nf * 16 + n] =
                f2bf(oacc[nf][v]);
}

// ---------------------------------------------------------------------------
// LayerNorm over width 512 (biased var, eps 1e-5), f32 in -> bf16 out.
// one block (256 threads) per row.
// ---------------------------------------------------------------------------
__global__ void ln_kernel(const float* __restrict__ X,
                          const float* __restrict__ g,
                          const float* __restrict__ b,
                          __bf16* __restrict__ Y) {
    const int row = blockIdx.x;
    const int tid = threadIdx.x;
    const float* xr = X + (size_t)row * 512;
    float x0 = xr[tid], x1 = xr[tid + 256];
    float s = x0 + x1, s2 = x0 * x0 + x1 * x1;
    for (int off = 16; off; off >>= 1) {
        s  += __shfl_down(s, off);
        s2 += __shfl_down(s2, off);
    }
    __shared__ float ws[8], ws2[8];
    if ((tid & 31) == 0) { ws[tid >> 5] = s; ws2[tid >> 5] = s2; }
    __syncthreads();
    if (tid == 0) {
        float a = 0.f, a2 = 0.f;
        for (int i = 0; i < 8; ++i) { a += ws[i]; a2 += ws2[i]; }
        ws[0] = a; ws2[0] = a2;
    }
    __syncthreads();
    float mean = ws[0] * (1.f / 512.f);
    float var  = ws2[0] * (1.f / 512.f) - mean * mean;
    float rstd = rsqrtf(var + 1e-5f);
    __bf16* yr = Y + (size_t)row * 512;
    yr[tid]       = f2bf((x0 - mean) * rstd * g[tid]       + b[tid]);
    yr[tid + 256] = f2bf((x1 - mean) * rstd * g[tid + 256] + b[tid + 256]);
}

// W[K][N] f32 -> Wt[N][K] bf16. grid*256 == K*N exactly.
__global__ void transpose_bf16(const float* __restrict__ W,
                               __bf16* __restrict__ Wt, int K, int N) {
    int idx = blockIdx.x * 256 + threadIdx.x;
    int n = idx / K, k = idx - n * K;
    Wt[idx] = f2bf(W[(size_t)k * N + n]);
}

// decoder query init + hi-res target. one block per (pb, s).
__global__ void embed_dec(const float* __restrict__ x,
                          const float* __restrict__ pos_w,
                          const float* __restrict__ pos_b,
                          const float* __restrict__ qtok,
                          float* __restrict__ o,
                          float* __restrict__ t) {
    int bx = blockIdx.x;               // pb*256 + s
    int pb = bx >> 8, s = bx & 255;
    int p = pb >> 3, b = pb & 7;
    int r1 = p >> 2, r2 = p & 3;
    int yy = (r1 << 4) + (s >> 4), xx = (r2 << 4) + (s & 15);
    float pxf = xx * (1.f / 64.f), pyf = yy * (1.f / 64.f);
    float* orow = o + (size_t)bx * 512;
    for (int d = threadIdx.x; d < 512; d += 128) {
        float v;
        if (d < 256) v = qtok[d];
        else { int dd = d - 256; v = pxf * pos_w[dd] + pyf * pos_w[256 + dd] + pos_b[dd]; }
        orow[d] = v;
    }
    if (threadIdx.x == 0) t[bx] = x[b * 4096 + yy * 64 + xx];
}

// encoder context init (bilinear 64x64 -> 4x4 samples coordinate 16i+7.5,
// i.e. the mean of the surrounding 2x2 pixels). one block per pb.
__global__ void embed_enc(const float* __restrict__ x,
                          const float* __restrict__ val_w,
                          const float* __restrict__ val_b,
                          const float* __restrict__ pos_w,
                          const float* __restrict__ pos_b,
                          float* __restrict__ m) {
    int pb = blockIdx.x;
    int p = pb >> 3, b = pb & 7;
    int py = p >> 2, px = p & 3;
    const float* xb = x + b * 4096;
    int y0 = py * 16 + 7, x0 = px * 16 + 7;
    float xlow = 0.25f * (xb[y0 * 64 + x0] + xb[y0 * 64 + x0 + 1] +
                          xb[(y0 + 1) * 64 + x0] + xb[(y0 + 1) * 64 + x0 + 1]);
    float pxf = px * 0.25f, pyf = py * 0.25f;
    float* mr = m + (size_t)pb * 512;
    for (int d = threadIdx.x; d < 512; d += 256) {
        float v;
        if (d < 256) v = xlow * val_w[d] + val_b[d];
        else { int dd = d - 256; v = pxf * pos_w[dd] + pyf * pos_w[256 + dd] + pos_b[dd]; }
        mr[d] = v;
    }
}

// o[row][:] += ca[row>>8][:]
__global__ void bcast_add(float* __restrict__ o, const float* __restrict__ ca) {
    int bx = blockIdx.x;
    const float* c = ca + (size_t)(bx >> 8) * 512;
    float* orow = o + (size_t)bx * 512;
    for (int d = threadIdx.x; d < 512; d += 128) orow[d] += c[d];
}

__global__ void zero_kernel(float* p) {
    if (threadIdx.x == 0 && blockIdx.x == 0) *p = 0.f;
}

// pred = ln(o) . dec_w + dec_b ; accumulate mean((pred - t)^2) into out.
__global__ void loss_kernel(const __bf16* __restrict__ lnb,
                            const float* __restrict__ dec_w,
                            const float* __restrict__ dec_b,
                            const float* __restrict__ t,
                            float* __restrict__ out) {
    int row = blockIdx.x, tid = threadIdx.x;  // 128 threads
    const __bf16* r = lnb + (size_t)row * 512;
    float s = 0.f;
    for (int d = tid; d < 512; d += 128) s += bf2f(r[d]) * dec_w[d];
    for (int off = 16; off; off >>= 1) s += __shfl_down(s, off);
    __shared__ float ws[4];
    if ((tid & 31) == 0) ws[tid >> 5] = s;
    __syncthreads();
    if (tid == 0) {
        float dot = ws[0] + ws[1] + ws[2] + ws[3] + dec_b[0];
        float diff = dot - t[row];
        atomicAdd(out, diff * diff * (1.0f / 32768.f));
    }
}

// ---------------------------------------------------------------------------
// Host orchestration
// ---------------------------------------------------------------------------
static const int  D512    = 512;
static const int  MROWS   = 32768;      // P*B*S_hi = 16*8*256
static const int  ATT_LDS = (256 * 64 + 64 * 256 + 4 * 16 * 256) * 2
                          + (4 * 16 * 256) * 4;   // 160KB

extern "C" void kernel_launch(void* const* d_in, const int* in_sizes, int n_in,
                              void* d_out, int out_size, void* d_ws, size_t ws_size,
                              hipStream_t stream) {
    (void)in_sizes; (void)n_in; (void)out_size; (void)ws_size;

    // ---- input mapping (setup_inputs() recursive insertion order) ----
    const float* X     = (const float*)d_in[0];
    const float* VAL_W = (const float*)d_in[1];
    const float* VAL_B = (const float*)d_in[2];
    const float* POS_W = (const float*)d_in[3];
    const float* POS_B = (const float*)d_in[4];
    const float* DEC_W = (const float*)d_in[5];
    const float* DEC_B = (const float*)d_in[6];
    const float* QTOK  = (const float*)d_in[7];
    // encoder params (L=4): sa_wq..sa_bo, ln1, ln2, w1,b1,w2,b2
    const float* E_WV  = (const float*)d_in[10];
    const float* E_WO  = (const float*)d_in[11];
    const float* E_BV  = (const float*)d_in[14];
    const float* E_BO  = (const float*)d_in[15];
    const float* E_LN1G= (const float*)d_in[16];
    const float* E_LN1B= (const float*)d_in[17];
    const float* E_LN2G= (const float*)d_in[18];
    const float* E_LN2B= (const float*)d_in[19];
    const float* E_W1  = (const float*)d_in[20];
    const float* E_B1  = (const float*)d_in[21];
    const float* E_W2  = (const float*)d_in[22];
    const float* E_B2  = (const float*)d_in[23];
    // decoder params (L=4)
    const float* D_WQ  = (const float*)d_in[24];
    const float* D_WK  = (const float*)d_in[25];
    const float* D_WV  = (const float*)d_in[26];
    const float* D_WO  = (const float*)d_in[27];
    const float* D_BQ  = (const float*)d_in[28];
    const float* D_BK  = (const float*)d_in[29];
    const float* D_BV  = (const float*)d_in[30];
    const float* D_BO  = (const float*)d_in[31];
    const float* C_WV  = (const float*)d_in[34];
    const float* C_WO  = (const float*)d_in[35];
    const float* C_BV  = (const float*)d_in[38];
    const float* C_BO  = (const float*)d_in[39];
    const float* D_LN3G= (const float*)d_in[40];
    const float* D_LN3B= (const float*)d_in[41];
    const float* D_LN1G= (const float*)d_in[42];
    const float* D_LN1B= (const float*)d_in[43];
    const float* D_W1  = (const float*)d_in[46];
    const float* D_B1  = (const float*)d_in[47];
    const float* D_W2  = (const float*)d_in[48];
    const float* D_B2  = (const float*)d_in[49];
    const float* FN_EG = (const float*)d_in[50];
    const float* FN_EB = (const float*)d_in[51];
    const float* FN_DG = (const float*)d_in[52];
    const float* FN_DB = (const float*)d_in[53];

    float* OUT = (float*)d_out;

    // ---- workspace carving ----
    char* cur = (char*)d_ws;
    auto alloc = [&](size_t bytes) -> char* {
        char* p = cur;
        cur += (bytes + 255) & ~(size_t)255;
        return p;
    };
    float*  o_buf   = (float*)alloc((size_t)MROWS * 512 * 4);
    float*  t_buf   = (float*)alloc((size_t)MROWS * 4);
    float*  m_buf   = (float*)alloc((size_t)128 * 512 * 4);
    float*  ca_buf  = (float*)alloc((size_t)128 * 512 * 4);
    __bf16* ln_buf  = (__bf16*)alloc((size_t)MROWS * 512 * 2);
    __bf16* qkv_buf = (__bf16*)alloc((size_t)MROWS * 1536 * 2);  // also FFN hidden
    __bf16* att_buf = (__bf16*)alloc((size_t)MROWS * 512 * 2);   // also enc V
    __bf16* mbf_buf = (__bf16*)alloc((size_t)128 * 512 * 2);
    __bf16* vmem_bf = (__bf16*)alloc((size_t)128 * 512 * 2);
    __bf16* h_buf   = qkv_buf;  // alias: qkv dead once attention finishes

    // bf16 transposed weights
    const size_t S55 = 512 * 512, S5T = 512 * 1024;
    __bf16 *eT_wv[4], *eT_wo[4], *eT_w1[4], *eT_w2[4];
    __bf16 *dT_wq[4], *dT_wk[4], *dT_wv[4], *dT_wo[4], *dT_cv[4], *dT_co[4], *dT_w1[4], *dT_w2[4];
    for (int i = 0; i < 4; ++i) {
        eT_wv[i] = (__bf16*)alloc(S55 * 2); eT_wo[i] = (__bf16*)alloc(S55 * 2);
        eT_w1[i] = (__bf16*)alloc(S5T * 2); eT_w2[i] = (__bf16*)alloc(S5T * 2);
        dT_wq[i] = (__bf16*)alloc(S55 * 2); dT_wk[i] = (__bf16*)alloc(S55 * 2);
        dT_wv[i] = (__bf16*)alloc(S55 * 2); dT_wo[i] = (__bf16*)alloc(S55 * 2);
        dT_cv[i] = (__bf16*)alloc(S55 * 2); dT_co[i] = (__bf16*)alloc(S55 * 2);
        dT_w1[i] = (__bf16*)alloc(S5T * 2); dT_w2[i] = (__bf16*)alloc(S5T * 2);
    }

    auto Tw = [&](const float* W, __bf16* Wt, int K, int N) {
        transpose_bf16<<<dim3((K * N) / 256), dim3(256), 0, stream>>>(W, Wt, K, N);
    };
    auto gemm_bf = [&](const __bf16* A, const __bf16* Wt, const float* bias,
                       __bf16* C, int ldc, int M, int N, int K) {
        gemm_kernel<false, false, true><<<dim3(M / 128, N / 64), dim3(128), 0, stream>>>(
            A, Wt, bias, nullptr, C, ldc, nullptr, K);
    };
    auto gemm_relu = [&](const __bf16* A, const __bf16* Wt, const float* bias,
                         __bf16* C, int ldc, int M, int N, int K) {
        gemm_kernel<true, false, true><<<dim3(M / 128, N / 64), dim3(128), 0, stream>>>(
            A, Wt, bias, nullptr, C, ldc, nullptr, K);
    };
    auto gemm_res = [&](const __bf16* A, const __bf16* Wt, const float* bias,
                        float* C, int ldc, const float* resid, int M, int N, int K) {
        gemm_kernel<false, true, false><<<dim3(M / 128, N / 64), dim3(128), 0, stream>>>(
            A, Wt, bias, C, nullptr, ldc, resid, K);
    };
    auto gemm_f32 = [&](const __bf16* A, const __bf16* Wt, const float* bias,
                        float* C, int ldc, int M, int N, int K) {
        gemm_kernel<false, false, false><<<dim3(M / 128, N / 64), dim3(128), 0, stream>>>(
            A, Wt, bias, C, nullptr, ldc, nullptr, K);
    };

    zero_kernel<<<dim3(1), dim3(32), 0, stream>>>(OUT);

    // ---- weight pre-pass (f32 -> bf16 transposed) ----
    for (int i = 0; i < 4; ++i) {
        Tw(E_WV + i * S55, eT_wv[i], 512, 512);
        Tw(E_WO + i * S55, eT_wo[i], 512, 512);
        Tw(E_W1 + i * S5T, eT_w1[i], 512, 1024);
        Tw(E_W2 + i * S5T, eT_w2[i], 1024, 512);
        Tw(D_WQ + i * S55, dT_wq[i], 512, 512);
        Tw(D_WK + i * S55, dT_wk[i], 512, 512);
        Tw(D_WV + i * S55, dT_wv[i], 512, 512);
        Tw(D_WO + i * S55, dT_wo[i], 512, 512);
        Tw(C_WV + i * S55, dT_cv[i], 512, 512);
        Tw(C_WO + i * S55, dT_co[i], 512, 512);
        Tw(D_W1 + i * S5T, dT_w1[i], 512, 1024);
        Tw(D_W2 + i * S5T, dT_w2[i], 1024, 512);
    }

    // ---- embeddings ----
    embed_enc<<<dim3(128), dim3(256), 0, stream>>>(X, VAL_W, VAL_B, POS_W, POS_B, m_buf);
    embed_dec<<<dim3(MROWS), dim3(128), 0, stream>>>(X, POS_W, POS_B, QTOK, o_buf, t_buf);

    // ---- encoder (seq len 1: attention == V exactly) ----
    for (int i = 0; i < 4; ++i) {
        ln_kernel<<<dim3(128), dim3(256), 0, stream>>>(m_buf, E_LN1G + i * D512, E_LN1B + i * D512, ln_buf);
        gemm_bf (ln_buf, eT_wv[i], E_BV + i * D512, att_buf, 512, 128, 512, 512);
        gemm_res(att_buf, eT_wo[i], E_BO + i * D512, m_buf, 512, m_buf, 128, 512, 512);
        ln_kernel<<<dim3(128), dim3(256), 0, stream>>>(m_buf, E_LN2G + i * D512, E_LN2B + i * D512, ln_buf);
        gemm_relu(ln_buf, eT_w1[i], E_B1 + i * 1024, h_buf, 1024, 128, 1024, 512);
        gemm_res (h_buf, eT_w2[i], E_B2 + i * D512, m_buf, 512, m_buf, 128, 512, 1024);
    }
    ln_kernel<<<dim3(128), dim3(256), 0, stream>>>(m_buf, FN_EG, FN_EB, mbf_buf);

    // ---- decoder ----
    for (int i = 0; i < 4; ++i) {
        // self-attention
        ln_kernel<<<dim3(MROWS), dim3(256), 0, stream>>>(o_buf, D_LN1G + i * D512, D_LN1B + i * D512, ln_buf);
        gemm_bf(ln_buf, dT_wq[i], D_BQ + i * D512, qkv_buf,        1536, MROWS, 512, 512);
        gemm_bf(ln_buf, dT_wk[i], D_BK + i * D512, qkv_buf + 512,  1536, MROWS, 512, 512);
        gemm_bf(ln_buf, dT_wv[i], D_BV + i * D512, qkv_buf + 1024, 1536, MROWS, 512, 512);
        attn_kernel<<<dim3(4, 8, 128), dim3(128), ATT_LDS, stream>>>(qkv_buf, att_buf);
        gemm_res(att_buf, dT_wo[i], D_BO + i * D512, o_buf, 512, o_buf, MROWS, 512, 512);
        // cross-attention: memory seq len == 1 -> output == V(memory), exactly
        gemm_bf (mbf_buf, dT_cv[i], C_BV + i * D512, vmem_bf, 512, 128, 512, 512);
        gemm_f32(vmem_bf, dT_co[i], C_BO + i * D512, ca_buf, 512, 128, 512, 512);
        bcast_add<<<dim3(MROWS), dim3(128), 0, stream>>>(o_buf, ca_buf);
        // FFN
        ln_kernel<<<dim3(MROWS), dim3(256), 0, stream>>>(o_buf, D_LN3G + i * D512, D_LN3B + i * D512, ln_buf);
        gemm_relu(ln_buf, dT_w1[i], D_B1 + i * 1024, h_buf, 1024, MROWS, 1024, 512);
        gemm_res (h_buf, dT_w2[i], D_B2 + i * D512, o_buf, 512, o_buf, MROWS, 512, 1024);
    }

    // ---- head + loss ----
    ln_kernel<<<dim3(MROWS), dim3(256), 0, stream>>>(o_buf, FN_DG, FN_DB, ln_buf);
    loss_kernel<<<dim3(MROWS), dim3(128), 0, stream>>>(ln_buf, DEC_W, DEC_B, t_buf, OUT);
}
